// TFEsmSelfAttention_47287589929371
// MI455X (gfx1250) — compile-verified, matrix operand order checked by reading, not run on hardware
//
#include <hip/hip_runtime.h>
#include <hip/hip_bf16.h>

#define B_  2
#define S_  2048
#define D_  1280
#define H_  20
#define HD_ 64

typedef __attribute__((ext_vector_type(16))) __bf16 v16bf;
typedef __attribute__((ext_vector_type(8)))  float  v8f;
typedef __attribute__((ext_vector_type(4)))  unsigned int v4u;
typedef __attribute__((ext_vector_type(8)))  int    v8i;
typedef __attribute__((ext_vector_type(4)))  int    v4i;

#if __has_builtin(__builtin_amdgcn_tensor_load_to_lds) && __has_builtin(__builtin_amdgcn_s_wait_tensorcnt)
#define HAVE_TDM 1
#else
#define HAVE_TDM 0
#endif

// ---------- helpers -------------------------------------------------------

__device__ __forceinline__ __bf16 f2bf(float f) {
    unsigned u = __float_as_uint(f);
    unsigned r = u + 0x7FFFu + ((u >> 16) & 1u);   // round-to-nearest-even
    unsigned short h = (unsigned short)(r >> 16);
    __bf16 o;
    __builtin_memcpy(&o, &h, 2);
    return o;
}

// Load a 16x32 bf16 WMMA A/B fragment for this lane.
// CDNA5 layout: lane L holds row/col (L%16); elements 0..7 = K {base8..base8+7},
// elements 8..15 = K {16+base8 .. 23+base8}, base8 = (L<16 ? 0 : 8).
// -> two 16-byte contiguous chunks (global_load_b128 / ds_load_b128).
__device__ __forceinline__ v16bf load_frag(const __bf16* p, int base8) {
    union { uint4 q[2]; v16bf v; } u;
    u.q[0] = *(const uint4*)(p + base8);
    u.q[1] = *(const uint4*)(p + base8 + 16);
    return u.v;
}

__device__ __forceinline__ v8f wmma_bf16(v16bf a, v16bf b, v8f c) {
    return __builtin_amdgcn_wmma_f32_16x16x32_bf16(
        /*neg_a=*/false, a, /*neg_b=*/false, b,
        /*c_mod=*/(short)0, c, /*reuse_a=*/false, /*reuse_b=*/false);
}

#if HAVE_TDM
// raw 32-bit LDS byte offset of a __shared__ object
__device__ __forceinline__ unsigned lds_off(const void* p) {
    return (unsigned)(unsigned long long)
        (__attribute__((address_space(3))) const void*)p;
}

// Build a 2D TDM D# (ISA ch.8) and issue TENSOR_LOAD_TO_LDS.
// Sizes/strides in bf16 elements (data_size code 1 = 2 bytes).
__device__ __forceinline__ void tdm_load_2d(
    const void* gptr, unsigned lds_byte_off,
    unsigned tile_d0, unsigned tile_d1,
    unsigned tensor_d0, unsigned tensor_d1,
    unsigned long long stride0)
{
    unsigned long long ga = (unsigned long long)gptr;
    v4u g0;
    g0[0] = 1u;                                          // count=1, user desc
    g0[1] = lds_byte_off;                                // lds_addr
    g0[2] = (unsigned)(ga & 0xFFFFFFFFu);                // global_addr[31:0]
    g0[3] = (unsigned)((ga >> 32) & 0x01FFFFFFu)         // global_addr[56:32]
            | 0x80000000u;                               // type=2 ("image")
    v8i g1;
    g1[0] = (int)(1u << 16);                             // data_size=1 (2B)
    g1[1] = (int)((tensor_d0 & 0xFFFFu) << 16);          // tensor_dim0[15:0]
    g1[2] = (int)(((tensor_d0 >> 16) & 0xFFFFu) |
                  ((tensor_d1 & 0xFFFFu) << 16));        // td0[31:16] | td1[15:0]
    g1[3] = (int)(((tensor_d1 >> 16) & 0xFFFFu) |
                  ((tile_d0 & 0xFFFFu) << 16));          // td1[31:16] | tile_d0
    g1[4] = (int)(tile_d1 & 0xFFFFu);                    // tile_d1; tile_d2=0
    g1[5] = (int)(unsigned)(stride0 & 0xFFFFFFFFu);      // dim0_stride[31:0]
    g1[6] = (int)(unsigned)((stride0 >> 32) & 0xFFFFu);  // dim0_stride[47:32]
    g1[7] = 0;                                           // dim1_stride unused (2D)
    v4i z = {0, 0, 0, 0};
#if __clang_major__ >= 23
    v8i z8 = {0, 0, 0, 0, 0, 0, 0, 0};
    __builtin_amdgcn_tensor_load_to_lds(g0, g1, z, z, z8, 0);   // clang-23: 6 args
#else
    __builtin_amdgcn_tensor_load_to_lds(g0, g1, z, z, 0);       // ROCm 7.2: 5 args
#endif
}
#endif

// ---------- kernel 0a: X fp32 -> bf16 ------------------------------------

__global__ void cvt_x(const float* __restrict__ X, __bf16* __restrict__ Xb) {
    size_t i = ((size_t)blockIdx.x * blockDim.x + threadIdx.x) * 4;
    float4 v = *(const float4*)(X + i);
    union { unsigned long long u; __bf16 e[4]; } pk;
    pk.e[0] = f2bf(v.x); pk.e[1] = f2bf(v.y);
    pk.e[2] = f2bf(v.z); pk.e[3] = f2bf(v.w);
    *(unsigned long long*)(Xb + i) = pk.u;
}

// ---------- kernel 0b: W fp32 [din][dout] -> bf16 Wt[m][dout][din] --------

__global__ void wtrans(const float* __restrict__ Wq, const float* __restrict__ Wk,
                       const float* __restrict__ Wv, __bf16* __restrict__ Wt) {
    size_t i = (size_t)blockIdx.x * blockDim.x + threadIdx.x;   // flat index in Wt
    int din  = (int)(i % D_);
    size_t t = i / D_;
    int dout = (int)(t % D_);
    int m    = (int)(t / D_);
    const float* W = (m == 0) ? Wq : ((m == 1) ? Wk : Wv);
    Wt[i] = f2bf(W[(size_t)din * D_ + dout]);
}

// ---------- kernel 1: QKV projection + bias + RoPE (+ Q scale) ------------
// One wave computes a 16(seq) x 64(one head) tile of one of Q/K/V.
// Q/K stored [B,H,S,64] bf16 (RoPE applied, Q pre-scaled by 1/8).
// V stored transposed [B,H,64,S] bf16.

__global__ __launch_bounds__(32) void qkv_rope(
    const __bf16* __restrict__ Xb, const __bf16* __restrict__ Wt,
    const float* __restrict__ bq, const float* __restrict__ bk,
    const float* __restrict__ bv,
    __bf16* __restrict__ Qb, __bf16* __restrict__ Kb, __bf16* __restrict__ Vt)
{
    int id = blockIdx.x;
    const int which = id % 3;        id /= 3;          // 0=Q 1=K 2=V
    const int h     = id % H_;       id /= H_;
    const int st    = id % (S_/16);  id /= (S_/16);
    const int b     = id;

    const int lane  = threadIdx.x;
    const int n     = lane & 15;
    const int base8 = (lane >> 4) * 8;

    const __bf16* xrow = Xb + (size_t)(b * S_ + st * 16 + n) * D_;
    const __bf16* wmat = Wt + (size_t)which * D_ * D_;
    const __bf16* wrow[4];
#pragma unroll
    for (int t = 0; t < 4; ++t)
        wrow[t] = wmat + (size_t)(h * HD_ + t * 16 + n) * D_;

    v8f acc[4] = {};
    for (int k = 0; k < D_; k += 32) {
        v16bf xa = load_frag(xrow + k, base8);
#pragma unroll
        for (int t = 0; t < 4; ++t)
            acc[t] = wmma_bf16(xa, load_frag(wrow[t] + k, base8), acc[t]);
    }

    const float* bias = (which == 0) ? bq : ((which == 1) ? bk : bv);
#pragma unroll
    for (int t = 0; t < 4; ++t) {
        float bt = bias[h * HD_ + t * 16 + n];
#pragma unroll
        for (int r = 0; r < 8; ++r) acc[t][r] += bt;
    }

    const int bh = b * H_ + h;

    if (which == 2) {
        // V: transpose-store. C-layout reg r holds seq row st*16 + base8 + r
        // at fixed dim d = t*16+n -> 8 consecutive bf16 = one b128 store.
#pragma unroll
        for (int t = 0; t < 4; ++t) {
            union { uint4 q; __bf16 e[8]; } pk;
#pragma unroll
            for (int r = 0; r < 8; ++r) pk.e[r] = f2bf(acc[t][r]);
            __bf16* dst = Vt + (size_t)(bh * HD_ + t * 16 + n) * S_
                             + st * 16 + base8;
            *(uint4*)dst = pk.q;
        }
    } else {
        const float scale = (which == 0) ? 0.125f : 1.0f;   // HD^-0.5
        __bf16* dstBase = ((which == 0) ? Qb : Kb) + (size_t)bh * S_ * HD_;
        const float LN1E4_OVER_32 = 0.2878231366242557f;    // ln(10000)/32
        float if0 = __expf(-(float)n        * LN1E4_OVER_32);
        float if1 = __expf(-(float)(16 + n) * LN1E4_OVER_32);
#pragma unroll
        for (int r = 0; r < 8; ++r) {
            int s = st * 16 + base8 + r;
            float th0 = (float)s * if0, th1 = (float)s * if1;
            float c0 = __cosf(th0), s0 = __sinf(th0);
            float c1 = __cosf(th1), s1 = __sinf(th1);
            float v0 = acc[0][r], v1 = acc[1][r];
            float v2 = acc[2][r], v3 = acc[3][r];
            float o0 = (v0 * c0 - v2 * s0) * scale;
            float o2 = (v2 * c0 + v0 * s0) * scale;
            float o1 = (v1 * c1 - v3 * s1) * scale;
            float o3 = (v3 * c1 + v1 * s1) * scale;
            __bf16* row = dstBase + (size_t)s * HD_ + n;
            row[0]  = f2bf(o0);
            row[16] = f2bf(o1);
            row[32] = f2bf(o2);
            row[48] = f2bf(o3);
        }
    }
}

// ---------- kernel 2: flash attention -------------------------------------
// 8 waves per block share one (b,h); wave w owns query tile qb*8+w.
// K/V^T tiles (32 keys) staged in LDS, double-buffered: wave 0 drives the
// Tensor Data Mover (TENSORcnt) so the DMA overlaps WMMA + softmax.
// Scores are computed transposed (S^T = K.Q^T) so the C-layout f32 registers
// remap to the P^T B-fragment with zero cross-lane traffic.

__device__ __forceinline__ void store8(float* p, v8f v, float rl) {
    float4 a = make_float4(v[0]*rl, v[1]*rl, v[2]*rl, v[3]*rl);
    float4 b = make_float4(v[4]*rl, v[5]*rl, v[6]*rl, v[7]*rl);
    *(float4*)p       = a;
    *(float4*)(p + 4) = b;
}

__global__ __launch_bounds__(256) void attn(
    const __bf16* __restrict__ Qb, const __bf16* __restrict__ Kb,
    const __bf16* __restrict__ Vt, float* __restrict__ out)
{
    __shared__ __align__(16) __bf16 Kt [2][32 * HD_];   // [keys][dims], 4KB each
    __shared__ __align__(16) __bf16 Vts[2][HD_ * 32];   // [dims][keys], 4KB each

    int id = blockIdx.x;
    const int qblk = id % (S_ / 128);  id /= (S_ / 128);
    const int h    = id % H_;          id /= H_;
    const int b    = id;
    const int bh   = b * H_ + h;

    const int wave  = threadIdx.x >> 5;
    const int lane  = threadIdx.x & 31;
    const int n     = lane & 15;
    const int base8 = (lane >> 4) * 8;
    const int qt    = qblk * 8 + wave;

    // Q as B-operand of S^T = K.Q^T: lane holds query column n, contiguous hd
    const __bf16* qrow = Qb + (size_t)(bh * S_ + qt * 16 + n) * HD_;
    v16bf qf0 = load_frag(qrow,      base8);
    v16bf qf1 = load_frag(qrow + 32, base8);

    const __bf16* kgbase = Kb + (size_t)bh * S_ * HD_;   // [S][64]
    const __bf16* vgbase = Vt + (size_t)bh * HD_ * S_;   // [64][S]

    v8f  o[4] = {};
    float mrun = -3.0e38f, lrun = 0.0f;

    const int NT = S_ / 32;

#if HAVE_TDM
    if (wave == 0) {   // prologue: stage tile 0 into buffer 0
        tdm_load_2d(kgbase, lds_off(&Kt[0][0]),  32 * HD_, 1, 32 * HD_, 1, 32 * HD_);
        tdm_load_2d(vgbase, lds_off(&Vts[0][0]), 32, HD_, S_, HD_, S_);
    }
#endif

#pragma unroll 1
    for (int it = 0; it < NT; ++it) {
        const int buf = it & 1;
#if HAVE_TDM
        if (wave == 0) {
            if (it + 1 < NT) {
                const int nb = buf ^ 1;
                tdm_load_2d(kgbase + (size_t)(it + 1) * 32 * HD_,
                            lds_off(&Kt[nb][0]), 32 * HD_, 1, 32 * HD_, 1, 32 * HD_);
                tdm_load_2d(vgbase + (size_t)(it + 1) * 32,
                            lds_off(&Vts[nb][0]), 32, HD_, S_, HD_, S_);
                __builtin_amdgcn_s_wait_tensorcnt(2);  // tile `it` complete
            } else {
                __builtin_amdgcn_s_wait_tensorcnt(0);
            }
        }
        __syncthreads();            // publish buf to all waves
#else
        __syncthreads();
        {   // cooperative fallback copy (256 threads x 16B per buffer)
            const int tid = threadIdx.x;
            *(uint4*)&Kt[buf][tid * 8] =
                *(const uint4*)(kgbase + (size_t)it * 32 * HD_ + tid * 8);
            const int col = (tid * 8) & 31, row = (tid * 8) >> 5;
            *(uint4*)&Vts[buf][row * 32 + col] =
                *(const uint4*)(vgbase + (size_t)row * S_ + it * 32 + col);
        }
        __syncthreads();
#endif

        // ---- scores transposed: keys (k0..k0+31) x 16 queries ----
        const __bf16* kr0 = &Kt[buf][(size_t)n * HD_];
        const __bf16* kr1 = &Kt[buf][(size_t)(n + 16) * HD_];
        v8f t0 = {}, t1 = {};
        t0 = wmma_bf16(load_frag(kr0,      base8), qf0, t0);
        t0 = wmma_bf16(load_frag(kr0 + 32, base8), qf1, t0);
        t1 = wmma_bf16(load_frag(kr1,      base8), qf0, t1);
        t1 = wmma_bf16(load_frag(kr1 + 32, base8), qf1, t1);

        float mx = t0[0];
#pragma unroll
        for (int r = 0; r < 8; ++r) { mx = fmaxf(mx, t0[r]); mx = fmaxf(mx, t1[r]); }
        mx = fmaxf(mx, __shfl_xor(mx, 16, 32));
        float mnew = fmaxf(mrun, mx);
        float corr = __expf(mrun - mnew);

        float p0[8], p1[8], sum = 0.0f;
#pragma unroll
        for (int r = 0; r < 8; ++r) {
            p0[r] = __expf(t0[r] - mnew);
            p1[r] = __expf(t1[r] - mnew);
            sum += p0[r] + p1[r];
        }
        sum += __shfl_xor(sum, 16, 32);
        lrun = lrun * corr + sum;
        mrun = mnew;

        // P^T B-fragment: elements 0..7 <- tile0 regs, 8..15 <- tile1 regs
        v16bf pf;
#pragma unroll
        for (int r = 0; r < 8; ++r) { pf[r] = f2bf(p0[r]); pf[8 + r] = f2bf(p1[r]); }

#pragma unroll
        for (int t = 0; t < 4; ++t) {
#pragma unroll
            for (int r = 0; r < 8; ++r) o[t][r] *= corr;
            o[t] = wmma_bf16(load_frag(&Vts[buf][(size_t)(t * 16 + n) * 32], base8),
                             pf, o[t]);
        }

        __syncthreads();            // all waves done reading buf before reuse
    }

    float rl = 1.0f / lrun;
    float* orow = out + (size_t)(b * S_ + qt * 16 + n) * D_ + h * HD_ + base8;
#pragma unroll
    for (int t = 0; t < 4; ++t) store8(orow + t * 16, o[t], rl);
}

// ---------- launch --------------------------------------------------------

extern "C" void kernel_launch(void* const* d_in, const int* in_sizes, int n_in,
                              void* d_out, int out_size, void* d_ws, size_t ws_size,
                              hipStream_t stream) {
    const float* X  = (const float*)d_in[0];
    const float* Wq = (const float*)d_in[1];
    const float* bq = (const float*)d_in[2];
    const float* Wk = (const float*)d_in[3];
    const float* bk = (const float*)d_in[4];
    const float* Wv = (const float*)d_in[5];
    const float* bv = (const float*)d_in[6];
    float* out = (float*)d_out;

    char* p = (char*)d_ws;
    const size_t szXb = (size_t)B_ * S_ * D_ * 2;        // 10.5 MB
    const size_t szWt = (size_t)3 * D_ * D_ * 2;         //  9.8 MB
    const size_t szQ  = (size_t)B_ * H_ * S_ * HD_ * 2;  // 10.5 MB each
    __bf16* Xb = (__bf16*)p; p += szXb;
    __bf16* Wt = (__bf16*)p; p += szWt;
    __bf16* Qb = (__bf16*)p; p += szQ;
    __bf16* Kb = (__bf16*)p; p += szQ;
    __bf16* Vt = (__bf16*)p; p += szQ;

    cvt_x   <<<dim3((B_ * S_ * D_) / (256 * 4)), 256, 0, stream>>>(X, Xb);
    wtrans  <<<dim3((3 * D_ * D_) / 256),        256, 0, stream>>>(Wq, Wk, Wv, Wt);
    qkv_rope<<<dim3(B_ * (S_/16) * H_ * 3),       32, 0, stream>>>(Xb, Wt, bq, bk, bv,
                                                                   Qb, Kb, Vt);
    attn    <<<dim3(B_ * H_ * (S_/128)),         256, 0, stream>>>(Qb, Kb, Vt, out);
}
